// ImprovedGlobalPointer_46780783788352
// MI455X (gfx1250) — matrix operand order repaired and checked by reading, not compiled, not run on hardware
//
#include <hip/hip_runtime.h>
#include <hip/hip_bf16.h>

// ---------------- problem constants ----------------
#define Bb   8
#define Ss   1024
#define HID  768
#define LEX  200
#define LEXP 224      // LEX padded to multiple of 32
#define ENT  9
#define Dd   64
#define ROWS (Bb*Ss)          // 8192
#define GKP  992              // 968 padded to multiple of 32
#define NOUT 1152             // ENT*2*D
#define QW   (ENT*Dd)         // 576
#define NEGC 1000000000000.0f

typedef __attribute__((ext_vector_type(16))) _Float16 v16h;
typedef __attribute__((ext_vector_type(8)))  float    v8f;

// ---- workspace layout (bytes) ----
#define OFF_HF    ((size_t)0)                          // 8192x768  f16
#define OFF_LEXP  (OFF_HF   + (size_t)ROWS*HID*2)      // 8192x224  f16
#define OFF_GW    (OFF_LEXP + (size_t)ROWS*LEXP*2)     // 992x768   f16
#define OFF_LW    (OFF_GW   + (size_t)GKP*HID*2)       // 224x768   f16
#define OFF_FUS   (OFF_LW   + (size_t)LEXP*HID*2)      // 8192x768  f16
#define OFF_DW    (OFF_FUS  + (size_t)ROWS*HID*2)      // 768x1152  f16
#define OFF_QF    (OFF_DW   + (size_t)HID*NOUT*2)      // 8192x576  f16
#define OFF_KT    (OFF_QF   + (size_t)ROWS*QW*2)       // 72x64x1024 f16

__device__ __forceinline__ v16h wmma_f16(v16h a, v16h b, v8f& c) {
  return __builtin_amdgcn_wmma_f32_16x16x32_f16(false, a, false, b, (short)0, c,
                                                false, false),
         a; // never used; see macro below
}
#define WMMA(a, b, c) \
  (c) = __builtin_amdgcn_wmma_f32_16x16x32_f16(false, (a), false, (b), (short)0, (c), false, false)

// Load one wave's A tile (16x32 f16) per ISA layout: lane holds row m0+lane%16;
// halves 0..7 = K kb..kb+7, halves 8..15 = K 16+kb..16+kb+7 (kb = (lane<16?0:8)).
__device__ __forceinline__ v16h load_a16(const _Float16* rowp, int kc, int kb) {
  union { v16h v; uint4 u[2]; } a;
  a.u[0] = *(const uint4*)(rowp + kc + kb);
  a.u[1] = *(const uint4*)(rowp + kc + 16 + kb);
  return a.v;
}

__device__ __forceinline__ v16h load_b16(const _Float16* p, int ld, int kb) {
  v16h b;
#pragma unroll
  for (int s = 0; s < 8; ++s) { b[s] = p[(size_t)(kb + s) * ld]; b[s + 8] = p[(size_t)(16 + kb + s) * ld]; }
  return b;
}

// ---------------- pack: f32 -> f16 with zero padding ----------------
__global__ void gp_pack_f16(const float* __restrict__ src, _Float16* __restrict__ dst,
                            int dR, int dC, int sR, int sC) {
  long long total = (long long)dR * dC;
  for (long long i = (long long)blockIdx.x * blockDim.x + threadIdx.x; i < total;
       i += (long long)gridDim.x * blockDim.x) {
    int r = (int)(i / dC), c = (int)(i % dC);
    float v = (r < sR && c < sC) ? src[(size_t)r * sC + c] : 0.0f;
    dst[i] = (_Float16)v;
  }
}

// ---------------- phase 1: gate/lex GEMMs + gated fuse ----------------
// one wave computes a 16x64 tile of fused (8192 x 768)
__global__ void gp_gate_fuse(const float* __restrict__ h32,
                             const _Float16* __restrict__ hF,
                             const _Float16* __restrict__ lxP,
                             const _Float16* __restrict__ gWf,
                             const _Float16* __restrict__ lWf,
                             const float* __restrict__ gate_b,
                             const float* __restrict__ lex_b,
                             _Float16* __restrict__ fused) {
  int wave = blockIdx.x * (blockDim.x >> 5) + (threadIdx.x >> 5);
  int lane = threadIdx.x & 31;
  int col  = lane & 15;
  int rsel = lane >> 4;
  int kb   = rsel * 8;
  int mt = wave / (HID / 64);      // 512 row tiles
  int nt = wave % (HID / 64);      // 12 col super-tiles
  int m0 = mt * 16, n0 = nt * 64;

  v8f accg[4] = {};
  v8f accl[4] = {};

  const _Float16* arow = hF + (size_t)(m0 + col) * HID;
  for (int kc = 0; kc < HID; kc += 32) {
    v16h a = load_a16(arow, kc, kb);
    const _Float16* bp = gWf + (size_t)kc * HID + n0 + col;
#pragma unroll
    for (int t = 0; t < 4; ++t) { v16h b = load_b16(bp + t * 16, HID, kb); WMMA(a, b, accg[t]); }
  }
  const _Float16* arl = lxP + (size_t)(m0 + col) * LEXP;
  for (int kc = 0; kc < LEXP; kc += 32) {
    v16h a = load_a16(arl, kc, kb);
    const _Float16* bg = gWf + (size_t)(HID + kc) * HID + n0 + col;
    const _Float16* bl = lWf + (size_t)kc * HID + n0 + col;
#pragma unroll
    for (int t = 0; t < 4; ++t) {
      v16h b0 = load_b16(bg + t * 16, HID, kb); WMMA(a, b0, accg[t]);
      v16h b1 = load_b16(bl + t * 16, HID, kb); WMMA(a, b1, accl[t]);
    }
  }

#pragma unroll
  for (int t = 0; t < 4; ++t) {
    int n = n0 + t * 16 + col;
    float gb = gate_b[n], lb = lex_b[n];
#pragma unroll
    for (int v = 0; v < 8; ++v) {
      int m = m0 + v + rsel * 8;
      float g  = 1.0f / (1.0f + __expf(-(accg[t][v] + gb)));
      float lh = tanhf(accl[t][v] + lb);
      float hv = h32[(size_t)m * HID + n];
      fused[(size_t)m * HID + n] = (_Float16)(g * lh + (1.0f - g) * hv);
    }
  }
}

// ---------------- phase 2: dense GEMM + RoPE, emit q and k^T ----------------
// one wave computes a 16x64 tile of out (8192 x 1152)
__global__ void gp_dense_rope(const _Float16* __restrict__ fused,
                              const _Float16* __restrict__ dWf,
                              const float* __restrict__ dense_b,
                              _Float16* __restrict__ qF,
                              _Float16* __restrict__ kT) {
  int wave = blockIdx.x * (blockDim.x >> 5) + (threadIdx.x >> 5);
  int lane = threadIdx.x & 31;
  int col  = lane & 15;
  int rsel = lane >> 4;
  int kb   = rsel * 8;
  int mt = wave / (NOUT / 64);     // 512
  int nt = wave % (NOUT / 64);     // 18
  int m0 = mt * 16, n0 = nt * 64;

  v8f acc[4] = {};
  const _Float16* arow = fused + (size_t)(m0 + col) * HID;
  for (int kc = 0; kc < HID; kc += 32) {
    v16h a = load_a16(arow, kc, kb);
    const _Float16* bp = dWf + (size_t)kc * NOUT + n0 + col;
#pragma unroll
    for (int t = 0; t < 4; ++t) { v16h b = load_b16(bp + t * 16, NOUT, kb); WMMA(a, b, acc[t]); }
  }

#pragma unroll
  for (int t = 0; t < 4; ++t) {
    int n  = n0 + t * 16 + col;
    int c  = n & 127;            // index within head (q:0..63, k:64..127)
    int hh = n >> 7;             // head
    int d  = c & 63;
    float db = dense_b[n];
    float invf = __expf(-(float)(d >> 1) * (9.210340371976184f / 32.0f)); // 10000^(-2i/64)
#pragma unroll
    for (int v = 0; v < 8; ++v) {
      int m = m0 + v + rsel * 8;
      float val = acc[t][v] + db;
      float partner = __shfl_xor(val, 1, 32);            // column n^1 == lane^1
      float rot = (n & 1) ? partner : -partner;          // rotate_half
      float pos = (float)(m & (Ss - 1));
      float se, ce;
      __sincosf(pos * invf, &se, &ce);
      float r = val * ce + rot * se;
      if (c < Dd) {
        qF[(size_t)m * QW + hh * Dd + d] = (_Float16)r;
      } else {
        int b = m >> 10, s = m & (Ss - 1);
        kT[(((size_t)(b * ENT + hh)) * Dd + d) * Ss + s] = (_Float16)r;
      }
    }
  }
}

// ---------------- phase 3: logits = q @ k^T, mask, scale ----------------
// one wave computes a 32x64 tile of one (b,h) 1024x1024 slab
__global__ void gp_logits(const _Float16* __restrict__ qF,
                          const _Float16* __restrict__ kT,
                          const float* __restrict__ amask,
                          float* __restrict__ out) {
  int wave = blockIdx.x * (blockDim.x >> 5) + (threadIdx.x >> 5);
  int lane = threadIdx.x & 31;
  int col  = lane & 15;
  int rsel = lane >> 4;
  int kb   = rsel * 8;
  int bh = wave >> 9;              // 512 waves per (b,h)
  int r  = wave & 511;
  int mt = r >> 4, nt = r & 15;
  int m0 = mt * 32, n0 = nt * 64;
  int b = bh / ENT, hh = bh % ENT;

  v8f acc[2][4] = {};
  const _Float16* kbase = kT + (size_t)bh * Dd * Ss;
  const _Float16* a0row = qF + (size_t)(b * Ss + m0 + col) * QW + hh * Dd;
  const _Float16* a1row = a0row + (size_t)16 * QW;
#pragma unroll
  for (int kc = 0; kc < Dd; kc += 32) {
    v16h a0 = load_a16(a0row, kc, kb);
    v16h a1 = load_a16(a1row, kc, kb);
    const _Float16* bp = kbase + (size_t)kc * Ss + n0 + col;
#pragma unroll
    for (int t = 0; t < 4; ++t) {
      v16h bm = load_b16(bp + t * 16, Ss, kb);
      WMMA(a0, bm, acc[0][t]);
      WMMA(a1, bm, acc[1][t]);
    }
  }

  float* obase = out + ((size_t)bh * Ss) * Ss;
#pragma unroll
  for (int half = 0; half < 2; ++half) {
#pragma unroll
    for (int t = 0; t < 4; ++t) {
      int n = n0 + t * 16 + col;
      float pad = amask[b * Ss + n];
      float maskoff = -(1.0f - pad) * NEGC;
#pragma unroll
      for (int v = 0; v < 8; ++v) {
        int m = m0 + half * 16 + v + rsel * 8;
        float x = acc[half][t][v] * pad + maskoff;
        if (n < m) x -= NEGC;                      // tril(ones,-1) causal term
        obase[(size_t)m * Ss + n] = x * 0.125f;    // 1/sqrt(64)
      }
    }
  }
}

extern "C" void kernel_launch(void* const* d_in, const int* in_sizes, int n_in,
                              void* d_out, int out_size, void* d_ws, size_t ws_size,
                              hipStream_t stream) {
  const float* h32     = (const float*)d_in[0];
  const float* lex32   = (const float*)d_in[1];
  const float* amask   = (const float*)d_in[2];
  const float* gateW32 = (const float*)d_in[3];
  const float* gate_b  = (const float*)d_in[4];
  const float* lexW32  = (const float*)d_in[5];
  const float* lex_b   = (const float*)d_in[6];
  const float* dnsW32  = (const float*)d_in[7];
  const float* dense_b = (const float*)d_in[8];
  float* out = (float*)d_out;

  char* ws = (char*)d_ws;
  _Float16* hF   = (_Float16*)(ws + OFF_HF);
  _Float16* lxP  = (_Float16*)(ws + OFF_LEXP);
  _Float16* gWf  = (_Float16*)(ws + OFF_GW);
  _Float16* lWf  = (_Float16*)(ws + OFF_LW);
  _Float16* fus  = (_Float16*)(ws + OFF_FUS);
  _Float16* dWf  = (_Float16*)(ws + OFF_DW);
  _Float16* qF   = (_Float16*)(ws + OFF_QF);
  _Float16* kT   = (_Float16*)(ws + OFF_KT);

  // pack/pad operands to f16
  gp_pack_f16<<<4096, 256, 0, stream>>>(h32,     hF,  ROWS, HID,  ROWS, HID);
  gp_pack_f16<<<2048, 256, 0, stream>>>(lex32,   lxP, ROWS, LEXP, ROWS, LEX);
  gp_pack_f16<<<1024, 256, 0, stream>>>(gateW32, gWf, GKP,  HID,  968,  HID);
  gp_pack_f16<<<256,  256, 0, stream>>>(lexW32,  lWf, LEXP, HID,  LEX,  HID);
  gp_pack_f16<<<1024, 256, 0, stream>>>(dnsW32,  dWf, HID,  NOUT, HID,  NOUT);

  // phase 1: 512*12 = 6144 waves, 8 waves/block
  gp_gate_fuse<<<768, 256, 0, stream>>>(h32, hF, lxP, gWf, lWf, gate_b, lex_b, fus);
  // phase 2: 512*18 = 9216 waves
  gp_dense_rope<<<1152, 256, 0, stream>>>(fus, dWf, dense_b, qF, kT);
  // phase 3: 72*512 = 36864 waves
  gp_logits<<<4608, 256, 0, stream>>>(qF, kT, amask, out);
}
